// MultiHeadAttention_43456479101334
// MI455X (gfx1250) — matrix operand order repaired and checked by reading, not compiled, not run on hardware
//
#include <hip/hip_runtime.h>

// ---------------------------------------------------------------------------
// MHA forward for MI455X (gfx1250, wave32, WMMA bf16 16x16x32, fp32 accum).
// Roofline: ~103 GFLOP vs ~650MB HBM traffic (512MB bias) -> memory-bound at
// bf16 matrix rates; all matmuls via v_wmma_f32_16x16x32_bf16.
// ---------------------------------------------------------------------------

#define B_SZ   8
#define HEADS  16
#define SEQ    1024
#define EMBED  1024
#define DK     64

typedef __attribute__((ext_vector_type(16))) __bf16 v16bf;
typedef __attribute__((ext_vector_type(8)))  float  v8f;

__device__ __forceinline__ unsigned short f2bf(float f) {
    unsigned int u = __float_as_uint(f);
    u += 0x7FFFu + ((u >> 16) & 1u);          // round-to-nearest-even
    return (unsigned short)(u >> 16);
}
__device__ __forceinline__ unsigned int pack2bf16(float lo, float hi) {
    unsigned int ul = __float_as_uint(lo); ul += 0x7FFFu + ((ul >> 16) & 1u);
    unsigned int uh = __float_as_uint(hi); uh += 0x7FFFu + ((uh >> 16) & 1u);
    return (ul >> 16) | (uh & 0xFFFF0000u);
}

union frag16 { unsigned int u[8]; v16bf v; };

// A fragment: 16x32 bf16, memory row-major [m*ld + k].
// lane l -> row m=l&15, half=l>>4; vgpr i<4: K=half*8+2i(,+1); i>=4: K=16+half*8+2(i-4).
__device__ __forceinline__ v16bf load_a_bf16(const unsigned short* base, int ld) {
    const int lane = threadIdx.x & 31;
    const int m = lane & 15, half = lane >> 4;
    const unsigned short* row = base + (size_t)m * ld;
    frag16 r;
#pragma unroll
    for (int i = 0; i < 8; ++i) {
        int kk = (i < 4) ? (half * 8 + 2 * i) : (16 + half * 8 + 2 * (i - 4));
        r.u[i] = *reinterpret_cast<const unsigned int*>(row + kk);
    }
    return r.v;
}
// Same, but source is fp32 (convert on the fly).
__device__ __forceinline__ v16bf load_a_f32(const float* base, int ld) {
    const int lane = threadIdx.x & 31;
    const int m = lane & 15, half = lane >> 4;
    const float* row = base + (size_t)m * ld;
    frag16 r;
#pragma unroll
    for (int i = 0; i < 8; ++i) {
        int kk = (i < 4) ? (half * 8 + 2 * i) : (16 + half * 8 + 2 * (i - 4));
        const float2 p = *reinterpret_cast<const float2*>(row + kk);
        r.u[i] = pack2bf16(p.x, p.y);
    }
    return r.v;
}
// B fragment: 32x16 bf16 B[k][n], memory stored so each output column's K values
// are contiguous: mem[n*ld + k]. lane l -> n=l&15, same K order as the A fragment.
__device__ __forceinline__ v16bf load_b_bf16(const unsigned short* base, int ld) {
    const int lane = threadIdx.x & 31;
    const int n = lane & 15, half = lane >> 4;
    const unsigned short* col = base + (size_t)n * ld;
    frag16 r;
#pragma unroll
    for (int i = 0; i < 8; ++i)
        r.u[i] = *reinterpret_cast<const unsigned int*>(col + ((i < 4) ? (half * 8 + 2 * i) : (16 + half * 8 + 2 * (i - 4))));
    return r.v;
}

__device__ __forceinline__ v8f wmma_bf16(v16bf a, v16bf b, v8f c) {
    return __builtin_amdgcn_wmma_f32_16x16x32_bf16(false, a, false, b, (short)0, c, false, false);
}

// ---------------------------------------------------------------------------
// Weight transpose + fp32->bf16:  WT[n*1024 + k] = bf16(W[k*1024 + n])
// ---------------------------------------------------------------------------
__global__ void wt_cvt_kernel(const float* __restrict__ W, unsigned short* __restrict__ WT) {
    int i = blockIdx.x * blockDim.x + threadIdx.x;      // 1M threads
    int n = i >> 10, k = i & 1023;
    WT[i] = f2bf(W[(size_t)k * EMBED + n]);
}

// ---------------------------------------------------------------------------
// GEMM: C[8192 x 1024] = A[8192 x 1024] @ W + bias.  One wave = 16x128 tile
// (8 WMMAs per 32-K step -> 8x A reuse per load).  Compile-time specialized:
// ABF16: 0 -> A is fp32 (convert on load), 1 -> A is bf16.
// OMODE: 0 -> bf16 to [B,H,N,DK]; 1 -> bf16 to [B,H,DK,N]; 2 -> fp32 row-major.
// ---------------------------------------------------------------------------
template <int ABF16, int OMODE>
__global__ __launch_bounds__(128) void gemm_proj_kernel(
    const void* __restrict__ Ain,
    const unsigned short* __restrict__ WT, const float* __restrict__ bvec,
    void* __restrict__ Cout)
{
    const int wave = threadIdx.x >> 5;
    const int lane = threadIdx.x & 31;
    const int gw = blockIdx.x * 4 + wave;      // 4096 waves
    const int col0 = (gw & 7) * 128;           // 8 column tiles of 128
    const int row0 = (gw >> 3) * 16;           // 512 row tiles of 16

    v8f acc[8] = {};
    for (int k0 = 0; k0 < EMBED; k0 += 32) {
        v16bf a;
        if (ABF16)
            a = load_a_bf16((const unsigned short*)Ain + (size_t)row0 * EMBED + k0, EMBED);
        else
            a = load_a_f32 ((const float*)Ain          + (size_t)row0 * EMBED + k0, EMBED);
#pragma unroll
        for (int j = 0; j < 8; ++j) {
            v16bf b = load_b_bf16(WT + (size_t)(col0 + j * 16) * EMBED + k0, EMBED);
            acc[j] = wmma_bf16(a, b, acc[j]);
        }
    }

    const int half = lane >> 4, nl = lane & 15;
#pragma unroll
    for (int j = 0; j < 8; ++j) {
        const int col = col0 + j * 16 + nl;
        const float bb = bvec[col];
#pragma unroll
        for (int r = 0; r < 8; ++r) {
            const int row = row0 + half * 8 + r;
            const float val = acc[j][r] + bb;
            if (OMODE == 2) {
                ((float*)Cout)[(size_t)row * EMBED + col] = val;
            } else {
                const int bb_ = row >> 10, n = row & 1023;   // batch, seq
                const int h   = col >> 6,  d = col & 63;     // head, dim
                const unsigned short bf = f2bf(val);
                if (OMODE == 0)
                    ((unsigned short*)Cout)[(((size_t)(bb_ * HEADS + h) * SEQ + n) << 6) + d] = bf;
                else
                    ((unsigned short*)Cout)[(((size_t)(bb_ * HEADS + h) * DK + d) << 10) + n] = bf;
            }
        }
    }
}

// ---------------------------------------------------------------------------
// Fused flash attention: one wave per (b, h, 16-row query tile).
// Online softmax over 32-key column tiles; 512MB bias streamed from HBM
// (the roofline-limiting term) with explicit prefetch of the next tile.
// Row-sum reduction is deferred: per-lane partials in-loop, one cross-lane
// reduction after the loop (halves ds_bpermute traffic vs naive).
// ---------------------------------------------------------------------------
__global__ __launch_bounds__(128) void attn_fused_kernel(
    const unsigned short* __restrict__ qh, const unsigned short* __restrict__ kh,
    const unsigned short* __restrict__ vhT, const float* __restrict__ bias,
    unsigned short* __restrict__ attn_out)
{
    __shared__ unsigned short plds[4][16 * 32];   // 1KB per wave: P tile staging

    const int wave = threadIdx.x >> 5;
    const int lane = threadIdx.x & 31;
    const int gw = blockIdx.x * 4 + wave;         // 8192 waves = 8*16*64
    const int rowtile = gw & 63;
    const int h = (gw >> 6) & 15;
    const int b = gw >> 10;

    const unsigned short* qhp = qh  + ((size_t)(b * HEADS + h) * SEQ + rowtile * 16) * DK;
    const unsigned short* khp = kh  + (size_t)(b * HEADS + h) * SEQ * DK;
    const unsigned short* vhp = vhT + (size_t)(b * HEADS + h) * DK * SEQ;
    const float* biasp = bias + (size_t)(b * HEADS + h) * SEQ * SEQ + (size_t)(rowtile * 16) * SEQ;

    const v16bf aq0 = load_a_bf16(qhp, DK);        // K = 0..31
    const v16bf aq1 = load_a_bf16(qhp + 32, DK);   // K = 32..63

    v8f O0 = {}, O1 = {}, O2 = {}, O3 = {};
    float m_run[8], l_part[8];                     // l_part: per-lane partial row sum
#pragma unroll
    for (int r = 0; r < 8; ++r) { m_run[r] = -1e30f; l_part[r] = 0.0f; }

    const int half = lane >> 4, nl = lane & 15;
    unsigned short* pw = &plds[wave][0];

    for (int c0 = 0; c0 < SEQ; c0 += 32) {
        // prefetch next iteration's bias tile (2 rows per lane covers 32 rows x 128B)
        if (c0 + 32 < SEQ) {
            const float* nb = biasp + c0 + 32;
            __builtin_prefetch(nb + (size_t)nl * SEQ + half * 16, 0, 0);
            __builtin_prefetch(nb + (size_t)(16 + nl) * SEQ + half * 16, 0, 0);
        }

        // ---- S = Q K^T : two 16x16 score fragments, K=64 in two WMMAs each
        v8f s0 = {}, s1 = {};
        s0 = wmma_bf16(aq0, load_b_bf16(khp + (size_t)c0 * DK, DK), s0);
        s0 = wmma_bf16(aq1, load_b_bf16(khp + (size_t)c0 * DK + 32, DK), s0);
        s1 = wmma_bf16(aq0, load_b_bf16(khp + (size_t)(c0 + 16) * DK, DK), s1);
        s1 = wmma_bf16(aq1, load_b_bf16(khp + (size_t)(c0 + 16) * DK + 32, DK), s1);

        // ---- scale + bias, tile row max
        float p0[8], p1[8], mx[8];
#pragma unroll
        for (int r = 0; r < 8; ++r) {
            const int row = half * 8 + r;
            p0[r] = s0[r] * 0.125f + biasp[(size_t)row * SEQ + c0 + nl];
            p1[r] = s1[r] * 0.125f + biasp[(size_t)row * SEQ + c0 + 16 + nl];
            mx[r] = fmaxf(p0[r], p1[r]);
        }
#pragma unroll
        for (int mask = 1; mask < 16; mask <<= 1)
#pragma unroll
            for (int r = 0; r < 8; ++r)
                mx[r] = fmaxf(mx[r], __shfl_xor(mx[r], mask, 32));

        // ---- online softmax update (per-lane partial sums, no in-loop reduce)
        float f[8];
#pragma unroll
        for (int r = 0; r < 8; ++r) {
            const float mn = fmaxf(m_run[r], mx[r]);
            f[r] = __expf(m_run[r] - mn);
            m_run[r] = mn;
            p0[r] = __expf(p0[r] - mn);
            p1[r] = __expf(p1[r] - mn);
            l_part[r] = l_part[r] * f[r] + p0[r] + p1[r];
            O0[r] *= f[r]; O1[r] *= f[r]; O2[r] *= f[r]; O3[r] *= f[r];
        }

        // ---- stage P (16x32 bf16) in LDS to re-layout D-frag -> A-frag
#pragma unroll
        for (int r = 0; r < 8; ++r) {
            const int row = half * 8 + r;
            pw[row * 32 + nl]      = f2bf(p0[r]);
            pw[row * 32 + 16 + nl] = f2bf(p1[r]);
        }
        asm volatile("s_wait_dscnt 0" ::: "memory");   // per-wave LDS RAW fence
        const v16bf pa = load_a_bf16(pw, 32);

        // ---- O += P @ V  (4 d-tiles of 16)
        O0 = wmma_bf16(pa, load_b_bf16(vhp + (size_t)0  * 1024 + c0, SEQ), O0);
        O1 = wmma_bf16(pa, load_b_bf16(vhp + (size_t)16 * 1024 + c0, SEQ), O1);
        O2 = wmma_bf16(pa, load_b_bf16(vhp + (size_t)32 * 1024 + c0, SEQ), O2);
        O3 = wmma_bf16(pa, load_b_bf16(vhp + (size_t)48 * 1024 + c0, SEQ), O3);
    }

    // ---- deferred row-sum reduction (once, after the loop)
#pragma unroll
    for (int mask = 1; mask < 16; mask <<= 1)
#pragma unroll
        for (int r = 0; r < 8; ++r)
            l_part[r] += __shfl_xor(l_part[r], mask, 32);

    // ---- normalize and emit [B, N, E] bf16 for the output projection
#pragma unroll
    for (int r = 0; r < 8; ++r) {
        const int row_g = rowtile * 16 + half * 8 + r;
        const float inv = 1.0f / l_part[r];
        const size_t base = ((size_t)(b * SEQ + row_g)) * EMBED + h * DK + nl;
        attn_out[base +  0] = f2bf(O0[r] * inv);
        attn_out[base + 16] = f2bf(O1[r] * inv);
        attn_out[base + 32] = f2bf(O2[r] * inv);
        attn_out[base + 48] = f2bf(O3[r] * inv);
    }
}

// ---------------------------------------------------------------------------
extern "C" void kernel_launch(void* const* d_in, const int* in_sizes, int n_in,
                              void* d_out, int out_size, void* d_ws, size_t ws_size,
                              hipStream_t stream) {
    const float* q    = (const float*)d_in[0];
    const float* k    = (const float*)d_in[1];
    const float* v    = (const float*)d_in[2];
    const float* bias = (const float*)d_in[3];
    const float* Wq   = (const float*)d_in[4];
    const float* bq   = (const float*)d_in[5];
    const float* Wv   = (const float*)d_in[6];
    const float* bv   = (const float*)d_in[7];
    const float* Wo   = (const float*)d_in[8];
    const float* bo   = (const float*)d_in[9];
    float* out = (float*)d_out;

    char* ws = (char*)d_ws;                                            // ~70MB used
    unsigned short* WqT = (unsigned short*)(ws + ((size_t)0  << 20));  // 2MB
    unsigned short* WvT = (unsigned short*)(ws + ((size_t)2  << 20));  // 2MB
    unsigned short* WoT = (unsigned short*)(ws + ((size_t)4  << 20));  // 2MB
    unsigned short* qh  = (unsigned short*)(ws + ((size_t)6  << 20));  // 16MB [B,H,N,DK]
    unsigned short* kh  = (unsigned short*)(ws + ((size_t)22 << 20));  // 16MB [B,H,N,DK]
    unsigned short* vhT = (unsigned short*)(ws + ((size_t)38 << 20));  // 16MB [B,H,DK,N]
    unsigned short* ao  = (unsigned short*)(ws + ((size_t)54 << 20));  // 16MB [B,N,E]

    // weights -> transposed bf16
    wt_cvt_kernel<<<4096, 256, 0, stream>>>(Wq, WqT);
    wt_cvt_kernel<<<4096, 256, 0, stream>>>(Wv, WvT);
    wt_cvt_kernel<<<4096, 256, 0, stream>>>(Wo, WoT);

    // projections (NOTE: k and v both use Wv/bv, matching the reference "bug")
    gemm_proj_kernel<0, 0><<<1024, 128, 0, stream>>>((const void*)q, WqT, bq, (void*)qh);
    gemm_proj_kernel<0, 0><<<1024, 128, 0, stream>>>((const void*)k, WvT, bv, (void*)kh);
    gemm_proj_kernel<0, 1><<<1024, 128, 0, stream>>>((const void*)v, WvT, bv, (void*)vhT);

    // fused attention with bias + online softmax
    attn_fused_kernel<<<2048, 128, 0, stream>>>(qh, kh, vhT, bias, ao);

    // output projection -> fp32
    gemm_proj_kernel<1, 2><<<1024, 128, 0, stream>>>((const void*)ao, WoT, bo, (void*)out);
}
